// CirLinear_40097814676033
// MI455X (gfx1250) — matrix operand order repaired and checked by reading, not compiled, not run on hardware
//
#include <hip/hip_runtime.h>

#define IN_F  2048
#define OUT_F 2048
#define M_TOT (16 * 1024)
#define NA    7

typedef __bf16 v16bf __attribute__((ext_vector_type(16)));
typedef float  v8f   __attribute__((ext_vector_type(8)));
typedef float  v4f   __attribute__((ext_vector_type(4)));

__device__ __forceinline__ unsigned short f2bf(float f) {
  unsigned int u = __float_as_uint(f);
  unsigned int rnd = ((u >> 16) & 1u) + 0x7FFFu;   // round-to-nearest-even
  return (unsigned short)((u + rnd) >> 16);
}

// ---------------------------------------------------------------------------
// Kernel X: convert x fp32 -> bf16 once.  x fp32 is read exactly once ->
// non-temporal loads; bf16 result is the GEMM working set -> regular stores.
// ---------------------------------------------------------------------------
__global__ void __launch_bounds__(256)
convert_x_kernel(const float* __restrict__ X, unsigned short* __restrict__ Xb) {
  size_t i = ((size_t)blockIdx.x * 256 + threadIdx.x) * 8;
  v4f a = __builtin_nontemporal_load(reinterpret_cast<const v4f*>(X + i));
  v4f b = __builtin_nontemporal_load(reinterpret_cast<const v4f*>(X + i + 4));
  union { unsigned short h[8]; uint4 q; } p;
  p.h[0] = f2bf(a.x); p.h[1] = f2bf(a.y); p.h[2] = f2bf(a.z); p.h[3] = f2bf(a.w);
  p.h[4] = f2bf(b.x); p.h[5] = f2bf(b.y); p.h[6] = f2bf(b.z); p.h[7] = f2bf(b.w);
  *reinterpret_cast<uint4*>(Xb + i) = p.q;
}

// ---------------------------------------------------------------------------
// Kernel A: build mixed circulant weight, bf16 [OUT_F][IN_F] into ws.
// One workgroup per 64x64 tile; all block sizes b<=64 are tile-local.
// ---------------------------------------------------------------------------
__global__ void __launch_bounds__(256)
build_weight_kernel(const float* __restrict__ W,
                    const float* __restrict__ alphas,
                    const float* __restrict__ gumbels,
                    unsigned short* __restrict__ wbf) {
  __shared__ float tile[64][68];
  __shared__ float sums[4032];
  __shared__ float aco[NA];

  const int tid = threadIdx.x;
  const int gI = blockIdx.x;
  const int gO = blockIdx.y;

  if (tid == 0) {
    float l[NA];
    float mx = -3.0e38f;
    for (int i = 0; i < NA; ++i) { l[i] = alphas[i] + gumbels[i]; mx = fmaxf(mx, l[i]); }
    float s = 0.f;
    for (int i = 0; i < NA; ++i) { l[i] = __expf(l[i] - mx); s += l[i]; }
    float inv = 1.f / s;
    for (int i = 0; i < NA; ++i) aco[i] = l[i] * inv;
  }

  const float* src = W + (size_t)(gO * 64) * IN_F + gI * 64;
  for (int i = 0; i < 4; ++i) {
    int e = tid + i * 256;
    int r = e >> 4, c4 = (e & 15) << 2;
    v4f v = __builtin_nontemporal_load(
        reinterpret_cast<const v4f*>(src + (size_t)r * IN_F + c4));
    tile[r][c4 + 0] = v.x; tile[r][c4 + 1] = v.y;
    tile[r][c4 + 2] = v.z; tile[r][c4 + 3] = v.w;
  }
  __syncthreads();

  const int BASES[6] = {0, 2048, 3072, 3584, 3840, 3968};
  #pragma unroll
  for (int lvl = 0; lvl < 6; ++lvl) {
    const int b = 2 << lvl;
    const int nsb = 64 / b;
    const int nsums = 4096 / b;
    for (int s = tid; s < nsums; s += 256) {
      int off = s & (b - 1);
      int blk = s / b;
      int sc = blk % nsb, sr = blk / nsb;
      int r0 = sr * b, c0 = sc * b;
      float acc = 0.f;
      for (int t = 0; t < b; ++t)
        acc += tile[r0 + t][c0 + ((t + off) & (b - 1))];
      sums[BASES[lvl] + s] = acc;
    }
  }
  __syncthreads();

  const float a0 = aco[0];
  for (int i = 0; i < 16; ++i) {
    int e = tid + i * 256;
    int r = e >> 6, c = e & 63;
    float val = a0 * tile[r][c];
    #pragma unroll
    for (int lvl = 0; lvl < 6; ++lvl) {
      const int b = 2 << lvl;
      const int nsb = 64 >> (lvl + 1);
      int off = (c - r) & (b - 1);
      int sr = r >> (lvl + 1), sc = c >> (lvl + 1);
      val += aco[lvl + 1] * (1.f / (float)b) *
             sums[BASES[lvl] + (sr * nsb + sc) * b + off];
    }
    wbf[(size_t)(gO * 64 + r) * IN_F + (gI * 64 + c)] = f2bf(val);
  }
}

// ---------------------------------------------------------------------------
// Kernel B: y = x * w^T + bias.  128x128 WG tile, 8 wave32 (2x4), each wave
// 4x2 WMMA 16x16x32 bf16 tiles.
// PRE=true : branch-free 3-stage async ring (GLOBAL_LOAD_ASYNC_TO_LDS_B128).
//            Every iteration issues one prefetch batch (K index wrapped at the
//            tail -> harmless dummy prefetch into an already-consumed slot)
//            and waits asynccnt<=4: everything except the newest batch is
//            complete, so stage kt+1 is always resident (in-order completion).
// PRE=false: 2-stage, x fp32 converted at the LDS store (small-ws fallback).
// #pragma unroll 1 keeps the accumulators in one fixed register set (no
// cross-body register rotation -> no v_mov copies, no WMMA hazard NOPs).
// ---------------------------------------------------------------------------
#define BM 128
#define BN 128
#define BK 32
#define LDA 40   // padded LDS row stride (elements) -> conflict-free b128

template <bool PRE>
__global__ void __launch_bounds__(256)
gemm_bias_kernel(const float* __restrict__ Xf,
                 const unsigned short* __restrict__ Xb,
                 const unsigned short* __restrict__ Wb,
                 const float* __restrict__ bias,
                 float* __restrict__ Y) {
  constexpr int NSTAGE = PRE ? 3 : 2;
  constexpr int LOOKAHEAD = PRE ? 2 : 1;
  constexpr int STAGE_ELEMS = BM * LDA;           // per-matrix stage size
  __shared__ unsigned short lA[NSTAGE][STAGE_ELEMS];
  __shared__ unsigned short lB[NSTAGE][STAGE_ELEMS];

  const int tid  = threadIdx.x;
  const int bn   = blockIdx.x;
  const int bm   = blockIdx.y;
  const int wave = tid >> 5;
  const int lane = tid & 31;
  const int lo   = lane & 15;
  const int hi   = lane >> 4;
  const int wm   = wave >> 2;
  const int wn   = wave & 3;

  const size_t xbase = (size_t)(bm * BM) * IN_F;
  const size_t wbase = (size_t)(bn * BN) * IN_F;

  v8f acc[4][2];
  #pragma unroll
  for (int i = 0; i < 4; ++i)
    #pragma unroll
    for (int j = 0; j < 2; ++j)
      #pragma unroll
      for (int r = 0; r < 8; ++r)
        acc[i][j][r] = 0.f;

  const unsigned long long xb64 = (unsigned long long)Xb;
  const unsigned long long wb64 = (unsigned long long)Wb;

  // load one K-stage into LDS slot at element offset `soff`
  auto load_stage = [&](int kt, int soff) {
    if constexpr (PRE) {
      // 512 x 16B chunks per matrix per stage; 2+2 async b128 per thread.
      #pragma unroll
      for (int i = 0; i < 2; ++i) {
        int idx = tid + i * 256;
        int r = idx >> 2;
        int kc = (idx & 3) << 3;
        unsigned la = (unsigned)(unsigned long long)&lA[0][soff + r * LDA + kc];
        unsigned oa = (unsigned)((xbase + (size_t)r * IN_F + kt * BK + kc) * 2);
        asm volatile("global_load_async_to_lds_b128 %0, %1, %2"
                     :: "v"(la), "v"(oa), "s"(xb64) : "memory");
        unsigned lb = (unsigned)(unsigned long long)&lB[0][soff + r * LDA + kc];
        unsigned ob = (unsigned)((wbase + (size_t)r * IN_F + kt * BK + kc) * 2);
        asm volatile("global_load_async_to_lds_b128 %0, %1, %2"
                     :: "v"(lb), "v"(ob), "s"(wb64) : "memory");
      }
    } else {
      #pragma unroll
      for (int i = 0; i < 4; ++i) {
        int idx = tid + i * 256;
        int r = idx >> 3;
        int kc = (idx & 7) << 2;
        float4 v = *reinterpret_cast<const float4*>(
            Xf + xbase + (size_t)r * IN_F + kt * BK + kc);
        union { unsigned short h[4]; unsigned long long u; } p;
        p.h[0] = f2bf(v.x); p.h[1] = f2bf(v.y);
        p.h[2] = f2bf(v.z); p.h[3] = f2bf(v.w);
        *reinterpret_cast<unsigned long long*>(&lA[0][soff + r * LDA + kc]) = p.u;
      }
      #pragma unroll
      for (int i = 0; i < 2; ++i) {
        int idx = tid + i * 256;
        int r = idx >> 2;
        int kc = (idx & 3) << 3;
        uint4 v = *reinterpret_cast<const uint4*>(
            Wb + wbase + (size_t)r * IN_F + kt * BK + kc);
        *reinterpret_cast<uint4*>(&lB[0][soff + r * LDA + kc]) = v;
      }
    }
  };

  // prologue: fill the first LOOKAHEAD stages
  load_stage(0, 0);
  if constexpr (PRE) {
    load_stage(1, STAGE_ELEMS);
    asm volatile("s_wait_asynccnt 0x4" ::: "memory");  // stage 0 resident
  }
  __syncthreads();

  const int NK = IN_F / BK;  // 64
  int soff = 0;                              // read-stage element offset
  int woff = LOOKAHEAD * STAGE_ELEMS;        // write-stage element offset
  if (woff >= NSTAGE * STAGE_ELEMS) woff -= NSTAGE * STAGE_ELEMS;

  #pragma unroll 1
  for (int kt = 0; kt < NK; ++kt) {
    // branch-free prefetch: wrap K index at the tail (harmless dummy prefetch
    // into a ring slot that was already consumed and is never read again).
    int pf = kt + LOOKAHEAD;
    if (pf >= NK) pf -= NK;
    load_stage(pf, woff);

    // A frags: lane lo = row M; halves 0..7 = K[hi*8..+7], 8..15 = K[16+hi*8..+7]
    v16bf af[4];
    #pragma unroll
    for (int mt = 0; mt < 4; ++mt) {
      const unsigned short* p = &lA[0][soff + (wm * 64 + mt * 16 + lo) * LDA];
      union { v16bf v; uint4 q[2]; } f;
      f.q[0] = *reinterpret_cast<const uint4*>(p + hi * 8);
      f.q[1] = *reinterpret_cast<const uint4*>(p + 16 + hi * 8);
      af[mt] = f.v;
    }
    // B frags: lane lo = col N; halves 0..7 = K[hi*8..+7], 8..15 = K[16+hi*8..+7] (w stored [N][K])
    v16bf bfr[2];
    #pragma unroll
    for (int nt = 0; nt < 2; ++nt) {
      const unsigned short* p = &lB[0][soff + (wn * 32 + nt * 16 + lo) * LDA];
      union { v16bf v; uint4 q[2]; } f;
      f.q[0] = *reinterpret_cast<const uint4*>(p + hi * 8);
      f.q[1] = *reinterpret_cast<const uint4*>(p + 16 + hi * 8);
      bfr[nt] = f.v;
    }

    #pragma unroll
    for (int mt = 0; mt < 4; ++mt)
      #pragma unroll
      for (int nt = 0; nt < 2; ++nt)
        acc[mt][nt] = __builtin_amdgcn_wmma_f32_16x16x32_bf16(
            false, af[mt], false, bfr[nt], (short)0, acc[mt][nt], false, false);

    if constexpr (PRE)
      asm volatile("s_wait_asynccnt 0x4" ::: "memory");  // next stage resident
    __syncthreads();

    soff += STAGE_ELEMS; if (soff == NSTAGE * STAGE_ELEMS) soff = 0;
    woff += STAGE_ELEMS; if (woff == NSTAGE * STAGE_ELEMS) woff = 0;
  }

  // epilogue: C VGPR r -> M = base + hi*8 + r, N = lo; bias add; NT stores
  // (y is write-once streaming data; keep x/w resident in L2).
  #pragma unroll
  for (int nt = 0; nt < 2; ++nt) {
    int n = bn * BN + wn * 32 + nt * 16 + lo;
    float bv = bias[n];
    #pragma unroll
    for (int mt = 0; mt < 4; ++mt) {
      int mbase = bm * BM + wm * 64 + mt * 16 + hi * 8;
      #pragma unroll
      for (int r = 0; r < 8; ++r)
        __builtin_nontemporal_store(acc[mt][nt][r] + bv,
                                    &Y[(size_t)(mbase + r) * OUT_F + n]);
    }
  }
}

// ---------------------------------------------------------------------------
extern "C" void kernel_launch(void* const* d_in, const int* in_sizes, int n_in,
                              void* d_out, int out_size, void* d_ws, size_t ws_size,
                              hipStream_t stream) {
  const float* x       = (const float*)d_in[0];
  const float* weight  = (const float*)d_in[1];
  const float* bias    = (const float*)d_in[2];
  const float* alphas  = (const float*)d_in[3];
  const float* gumbels = (const float*)d_in[4];
  float* out = (float*)d_out;

  unsigned short* wbf = (unsigned short*)d_ws;             // 8 MB
  const size_t needW = (size_t)OUT_F * IN_F * 2;
  const size_t needX = (size_t)M_TOT * IN_F * 2;           // 64 MB

  dim3 gA(IN_F / 64, OUT_F / 64);
  build_weight_kernel<<<gA, 256, 0, stream>>>(weight, alphas, gumbels, wbf);

  dim3 gB(OUT_F / BN, M_TOT / BM);
  if (ws_size >= needW + needX) {
    unsigned short* xbf = wbf + (size_t)OUT_F * IN_F;
    convert_x_kernel<<<(M_TOT * IN_F) / (256 * 8), 256, 0, stream>>>(x, xbf);
    gemm_bias_kernel<true><<<gB, 256, 0, stream>>>(nullptr, xbf, wbf, bias, out);
  } else {
    gemm_bias_kernel<false><<<gB, 256, 0, stream>>>(x, nullptr, wbf, bias, out);
  }
}
